// PCDAlignment_55637006353025
// MI455X (gfx1250) — matrix operand order repaired and checked by reading, not compiled, not run on hardware
//
#include <hip/hip_runtime.h>

typedef _Float16 half_t;
typedef _Float16 v16h __attribute__((ext_vector_type(16)));
typedef float    v8f  __attribute__((ext_vector_type(8)));

#define NFEAT 64
#define OCH   144   // offset channels = 2*9*8

// ---------------------------------------------------------------------------
// Repack fp32 OIHW conv weights into f16 WMMA A-fragments.
// GEMM K-ordering is kernel-position-major with channel padding:
//     kf = kk*Cpad + c,  kk = ky*3+kx in [0,9),  c in [0,Cpad),  Cpad = ceil32(Cin)
// Channels c >= Cin are zero-padded, so the conv kernel never needs a K tail.
// Fragment layout: frag[((tile*ksteps + ks)*32 + lane)*16 + j], with the
// A-matrix 16x32 f16 layout (contiguous K order): lane m (0-15) holds K {0-15},
// lane m+16 holds K {16-31}; element j -> K = 16*hi + j.
// ---------------------------------------------------------------------------
__global__ __launch_bounds__(128)
void pack_weights_kernel(const float* __restrict__ w, half_t* __restrict__ pk,
                         int Cin, int Cpad, int ksteps, int total)
{
    int t = blockIdx.x * blockDim.x + threadIdx.x;   // one A-fragment per thread
    if (t >= total) return;
    int lane = t & 31;
    int ks   = (t >> 5) % ksteps;
    int tile = (t >> 5) / ksteps;
    int m  = lane & 15;
    int hi = lane >> 4;
    v16h frag;
    #pragma unroll
    for (int j = 0; j < 16; ++j) {
        int klocal = j + (hi << 4);
        int kf = (ks << 5) + klocal;
        int kk = kf / Cpad;
        int c  = kf - kk * Cpad;
        float v = (c < Cin) ? w[(tile * 16 + m) * (Cin * 9) + c * 9 + kk] : 0.f;
        frag[j] = (half_t)v;
    }
    *(v16h*)(pk + (size_t)t * 16) = frag;
}

// ---------------------------------------------------------------------------
// Implicit-GEMM 3x3 conv (stride 1, pad 1) via V_WMMA_F32_16X16X32_F16.
// One wave32 per 16(outch) x 16(pixel) tile; pixels are 16 consecutive x.
// Templated on H,W so the 16 channel-strided B loads per K-step use
// compile-time immediate offsets (j*HW*4 fits the 24-bit IOFFSET).
// Input is a virtual concat of two NCHW tensors (C1 + C2 channels); a
// 16-channel run never straddles the C1 boundary (C1 multiple of 16 and the
// hi/lo K-half split is 16-channel aligned). The concat select is a plain
// value select between two precomputed row pointers (no divergent branch).
// ---------------------------------------------------------------------------
template<int H, int W>
__global__ __launch_bounds__(32)
void conv3x3_wmma_kernel(const float* __restrict__ in1, int C1,
                         const float* __restrict__ in2, int C2,
                         const half_t* __restrict__ pk,
                         const float* __restrict__ bias,
                         float* __restrict__ out, int Cout, int leaky)
{
    constexpr int HW = H * W;
    const int lane = threadIdx.x;            // 0..31
    const int Cin  = C1 + C2;
    const int CF   = Cin >> 5;               // full 32-channel blocks
    const bool hasPartial = (Cin & 31) != 0; // only Cin==144 (uniform)
    const int p0   = blockIdx.x * 16;        // pixel tile base (16 consecutive x)
    const int tile = blockIdx.y;             // out-channel tile

    const int n    = lane & 15;              // B column / pixel
    const int pn   = p0 + n;
    const int b    = pn / HW;
    const int rem  = pn - b * HW;
    const int y    = rem / W;
    const int x    = rem - y * W;
    const int hi16 = (lane >> 4) << 4;       // K half owned by this lane

    const float* base1 = in1 + b * C1 * HW;
    const float* base2v = in2 ? (in2 + b * C2 * HW) : (in1 + b * C1 * HW);
    // shifted so that (base2 + cu*HW) with the *global* channel index works
    const float* base2 = base2v - (ptrdiff_t)C1 * HW;

    const half_t* pkl = pk + ((size_t)tile * (9 * (hasPartial ? CF + 1 : CF)) * 32
                              + lane) * 16;

    v8f acc = {0.f, 0.f, 0.f, 0.f, 0.f, 0.f, 0.f, 0.f};

    int s = 0;
    for (int kk = 0; kk < 9; ++kk) {
        const int ky = kk / 3 - 1;
        const int kx = (kk - (ky + 1) * 3) - 1;
        const int sy = y + ky;
        const int sx = x + kx;
        const bool oksp = ((unsigned)sy < (unsigned)H) && ((unsigned)sx < (unsigned)W);
        const int syc = sy < 0 ? 0 : (sy >= H ? H - 1 : sy);
        const int sxc = sx < 0 ? 0 : (sx >= W ? W - 1 : sx);
        const int sp  = syc * W + sxc;
        const float* prow1 = base1 + sp;     // per-(ky,kx) row pointers
        const float* prow2 = base2 + sp;

        for (int cb = 0; cb < CF; ++cb, ++s) {   // full blocks: spatial mask only
            v16h a = *(const v16h*)(pkl + (size_t)s * 512);
            const int cu = (cb << 5) + hi16;     // first channel of 16-run
            const float* p = ((cu < C1) ? prow1 : prow2) + cu * HW;
            v16h bm;
            #pragma unroll
            for (int j = 0; j < 16; ++j) {       // 16 loads, imm offsets
                float v = p[j * HW];
                bm[j] = oksp ? (half_t)v : (half_t)0.f;
            }
            acc = __builtin_amdgcn_wmma_f32_16x16x32_f16(
                      false, a, false, bm, (short)0, acc, false, false);
        }
        if (hasPartial) {                        // uniform branch, EXEC full
            v16h a = *(const v16h*)(pkl + (size_t)s * 512);
            const int  cu = (CF << 5) + hi16;
            const bool ok = oksp && (cu < Cin);
            const int  cc = (cu < Cin) ? cu : 0; // clamp for address safety
            const float* p = ((cc < C1) ? prow1 : prow2) + cc * HW;
            v16h bm;
            #pragma unroll
            for (int j = 0; j < 16; ++j) {
                float v = p[j * HW];
                bm[j] = ok ? (half_t)v : (half_t)0.f;
            }
            acc = __builtin_amdgcn_wmma_f32_16x16x32_f16(
                      false, a, false, bm, (short)0, acc, false, false);
            ++s;
        }
    }

    // C/D layout: VGPR r -> M=r (lanes 0-15) / M=r+8 (lanes 16-31), N=lane&15
    #pragma unroll
    for (int r = 0; r < 8; ++r) {
        int oc = tile * 16 + r + (hi16 >> 1);   // hi16>>1 == hi*8
        float v = acc[r];
        if (bias)  v += bias[oc];
        if (leaky) v = (v >= 0.f) ? v : 0.1f * v;
        out[((b * Cout + oc) * H + y) * W + x] = v;
    }
}

// ---------------------------------------------------------------------------
// DCN v1 (3x3, pad 1, 8 deformable groups, Cg=8, no modulation, no bias).
// One 256-thread block per 16-pixel tile:
//   phase 1: cooperatively bilinear-sample s[pix][cin*9+k] into LDS
//   phase 2: 64 out-channels x 4 pixels per thread, float4 weights + LDS MAC
// ---------------------------------------------------------------------------
#define SSTR 584   // 576 padded (+8) to break 64-bank alignment across pixels
template<int H, int W>
__global__ __launch_bounds__(256)
void dcn16_kernel(const float* __restrict__ in,
                  const float* __restrict__ off,
                  const float* __restrict__ w,
                  float* __restrict__ out, int leaky)
{
    constexpr int HW = H * W;
    __shared__ float s[16 * SSTR];
    const int tile = blockIdx.x * 16;
    const int tid  = threadIdx.x;

    // phase 1: 16*576 = 9216 samples, 36 per thread (branchless taps)
    for (int it = 0; it < 36; ++it) {
        int idx = tid + (it << 8);
        int pix = idx / 576;
        int r   = idx - pix * 576;
        int cin = r / 9;
        int k   = r - cin * 9;
        int g   = cin >> 3;
        int pn  = tile + pix;
        int b   = pn / HW;
        int rem = pn - b * HW;
        int y   = rem / W;
        int x   = rem - y * W;
        int ky  = k / 3 - 1;
        int kx  = (k - (k / 3) * 3) - 1;

        int obase = ((b * OCH + (g * 18 + k * 2)) * H + y) * W + x;
        float dy = off[obase];          // channel g*18 + k*2 + 0
        float dx = off[obase + HW];     // channel g*18 + k*2 + 1
        float py = dy + (float)(ky + y);
        float px = dx + (float)(kx + x);
        float fy = floorf(py), fx = floorf(px);
        float wy = py - fy,    wx = px - fx;
        int y0 = (int)fy, x0 = (int)fx;
        int y1 = y0 + 1,  x1 = x0 + 1;

        bool yi0 = (unsigned)y0 < (unsigned)H;
        bool yi1 = (unsigned)y1 < (unsigned)H;
        bool xi0 = (unsigned)x0 < (unsigned)W;
        bool xi1 = (unsigned)x1 < (unsigned)W;
        int y0c = y0 < 0 ? 0 : (y0 >= H ? H - 1 : y0);
        int y1c = y1 < 0 ? 0 : (y1 >= H ? H - 1 : y1);
        int x0c = x0 < 0 ? 0 : (x0 >= W ? W - 1 : x0);
        int x1c = x1 < 0 ? 0 : (x1 >= W ? W - 1 : x1);

        const float* img = in + (b * NFEAT + cin) * HW;
        float t00 = img[y0c * W + x0c]; t00 = (yi0 && xi0) ? t00 : 0.f;
        float t01 = img[y0c * W + x1c]; t01 = (yi0 && xi1) ? t01 : 0.f;
        float t10 = img[y1c * W + x0c]; t10 = (yi1 && xi0) ? t10 : 0.f;
        float t11 = img[y1c * W + x1c]; t11 = (yi1 && xi1) ? t11 : 0.f;
        s[pix * SSTR + r] = t00 * (1.f - wy) * (1.f - wx)
                          + t01 * (1.f - wy) * wx
                          + t10 * wy * (1.f - wx)
                          + t11 * wy * wx;
    }
    __syncthreads();

    // phase 2
    const int oc = tid & 63;
    const int pg = tid >> 6;                  // 0..3 -> pixels pg*4 .. pg*4+3
    const float4* wrow4 = (const float4*)(w + oc * 576); // o*576 + cin*9 + k
    const float* sp0 = s + (pg * 4 + 0) * SSTR;
    const float* sp1 = s + (pg * 4 + 1) * SSTR;
    const float* sp2 = s + (pg * 4 + 2) * SSTR;
    const float* sp3 = s + (pg * 4 + 3) * SSTR;
    float a0 = 0.f, a1 = 0.f, a2 = 0.f, a3 = 0.f;
    #pragma unroll 4
    for (int r4 = 0; r4 < 144; ++r4) {
        float4 wv = wrow4[r4];
        int r = r4 << 2;
        a0 += wv.x * sp0[r] + wv.y * sp0[r + 1] + wv.z * sp0[r + 2] + wv.w * sp0[r + 3];
        a1 += wv.x * sp1[r] + wv.y * sp1[r + 1] + wv.z * sp1[r + 2] + wv.w * sp1[r + 3];
        a2 += wv.x * sp2[r] + wv.y * sp2[r + 1] + wv.z * sp2[r + 2] + wv.w * sp2[r + 3];
        a3 += wv.x * sp3[r] + wv.y * sp3[r + 1] + wv.z * sp3[r + 2] + wv.w * sp3[r + 3];
    }
    float accs[4] = {a0, a1, a2, a3};
    #pragma unroll
    for (int i = 0; i < 4; ++i) {
        int pn  = tile + pg * 4 + i;
        int b   = pn / HW;
        int rem = pn - b * HW;
        int y   = rem / W;
        int x   = rem - y * W;
        float v = accs[i];
        if (leaky) v = (v >= 0.f) ? v : 0.1f * v;
        out[((b * NFEAT + oc) * H + y) * W + x] = v;
    }
}

// ---------------------------------------------------------------------------
// Bilinear 2x upsample (align_corners=False, half-pixel, edge clamp),
// with optional output scale (x2 for offset fields).
// ---------------------------------------------------------------------------
__global__ __launch_bounds__(256)
void upsample2x_kernel(const float* __restrict__ in, float* __restrict__ out,
                       int C, int H, int W, float scale, int total)
{
    int idx = blockIdx.x * blockDim.x + threadIdx.x;
    if (idx >= total) return;
    const int Wo = 2 * W, Ho = 2 * H;
    int xo = idx % Wo;  int t = idx / Wo;
    int yo = t % Ho;    t /= Ho;
    int c  = t % C;     int b = t / C;

    float sy = (yo + 0.5f) * 0.5f - 0.5f;
    float sx = (xo + 0.5f) * 0.5f - 0.5f;
    float fy = floorf(sy), fx = floorf(sx);
    float wy = sy - fy,    wx = sx - fx;
    int y0 = (int)fy, x0 = (int)fx;
    int y0c = y0 < 0 ? 0 : (y0 > H - 1 ? H - 1 : y0);
    int y1t = y0 + 1;
    int y1c = y1t < 0 ? 0 : (y1t > H - 1 ? H - 1 : y1t);
    int x0c = x0 < 0 ? 0 : (x0 > W - 1 ? W - 1 : x0);
    int x1t = x0 + 1;
    int x1c = x1t < 0 ? 0 : (x1t > W - 1 ? W - 1 : x1t);

    const float* img = in + (b * C + c) * H * W;
    float v = img[y0c * W + x0c] * (1.f - wy) * (1.f - wx)
            + img[y0c * W + x1c] * (1.f - wy) * wx
            + img[y1c * W + x0c] * wy * (1.f - wx)
            + img[y1c * W + x1c] * wy * wx;
    out[idx] = v * scale;
}

// ---------------------------------------------------------------------------
// Orchestration (B = 4, pyramid 128/64/32)
// ---------------------------------------------------------------------------
extern "C" void kernel_launch(void* const* d_in, const int* in_sizes, int n_in,
                              void* d_out, int out_size, void* d_ws, size_t ws_size,
                              hipStream_t stream)
{
    (void)in_sizes; (void)n_in; (void)out_size; (void)ws_size;

    const float* nbr1 = (const float*)d_in[0];
    const float* ref1 = (const float*)d_in[1];
    const float* nbr2 = (const float*)d_in[2];
    const float* ref2 = (const float*)d_in[3];
    const float* nbr3 = (const float*)d_in[4];
    const float* ref3 = (const float*)d_in[5];

    // params in dict insertion order
    const float* P[28];
    for (int i = 0; i < 28; ++i) P[i] = (const float*)d_in[6 + i];
    // l=0: 0 oc1_w0, 1 oc1_b0, 2 oc2_w0, 3 oc2_b0, 4 oc3_w0, 5 oc3_b0,
    //      6 fc_w0, 7 fc_b0, 8 dcn_w0
    // l=1: 9..17 (same layout)
    // l=2: 18 oc1_w2, 19 oc1_b2, 20 oc2_w2, 21 oc2_b2, 22 dcn_w2
    // cas: 23 cas1_w, 24 cas1_b, 25 cas2_w, 26 cas2_b, 27 casdcn_w

    // workspace layout (floats), all chunks multiples of 8 -> 32B alignment ok
    const int OFF_SZ  = 4 * OCH   * 128 * 128;   // 9,437,184 floats
    const int FEAT_SZ = 4 * NFEAT * 128 * 128;   // 4,194,304 floats
    float*  ws     = (float*)d_ws;
    float*  offA   = ws;
    float*  offB   = offA   + OFF_SZ;
    float*  upOff  = offB   + OFF_SZ;
    float*  featD  = upOff  + OFF_SZ;
    float*  featE  = featD  + FEAT_SZ;
    float*  upFeat = featE  + FEAT_SZ;
    half_t* pk     = (half_t*)(upFeat + FEAT_SZ);  // packed weights (<=746KB)

    auto conv = [stream, pk](const float* i1, int c1, const float* i2, int c2,
                             const float* wt, const float* bs,
                             float* o, int cout, int H, int W, int lk) {
        int cin    = c1 + c2;
        int cpad   = (cin + 31) & ~31;
        int ksteps = 9 * (cpad >> 5);
        int total  = (cout / 16) * ksteps * 32;
        pack_weights_kernel<<<dim3((unsigned)((total + 127) / 128)), dim3(128),
                              0, stream>>>(wt, pk, cin, cpad, ksteps, total);
        dim3 grid((unsigned)((4 * H * W) / 16), (unsigned)(cout / 16));
        if (H == 128)
            conv3x3_wmma_kernel<128, 128><<<grid, dim3(32), 0, stream>>>(
                i1, c1, i2, c2, pk, bs, o, cout, lk);
        else if (H == 64)
            conv3x3_wmma_kernel<64, 64><<<grid, dim3(32), 0, stream>>>(
                i1, c1, i2, c2, pk, bs, o, cout, lk);
        else
            conv3x3_wmma_kernel<32, 32><<<grid, dim3(32), 0, stream>>>(
                i1, c1, i2, c2, pk, bs, o, cout, lk);
    };
    auto dcn = [stream](const float* in, const float* off, const float* wt,
                        float* o, int H, int W, int lk) {
        dim3 grid((unsigned)((4 * H * W) / 16));
        if (H == 128)
            dcn16_kernel<128, 128><<<grid, dim3(256), 0, stream>>>(in, off, wt, o, lk);
        else if (H == 64)
            dcn16_kernel<64, 64><<<grid, dim3(256), 0, stream>>>(in, off, wt, o, lk);
        else
            dcn16_kernel<32, 32><<<grid, dim3(256), 0, stream>>>(in, off, wt, o, lk);
    };
    auto up = [stream](const float* in, float* o, int C, int H, int W, float sc) {
        int total = 4 * C * 4 * H * W;
        upsample2x_kernel<<<dim3((unsigned)((total + 255) / 256)), dim3(256), 0, stream>>>(
            in, o, C, H, W, sc, total);
    };

    // ---------------- Level 3 (32x32) ----------------
    conv(nbr3, 64, ref3, 64, P[18], P[19], offA, OCH, 32, 32, 1);   // oc1
    conv(offA, OCH, nullptr, 0, P[20], P[21], offB, OCH, 32, 32, 1);// oc2
    dcn(nbr3, offB, P[22], featD, 32, 32, 1);                       // feat = leaky(dcn)
    up(offB,  upOff,  OCH,   32, 32, 2.0f);                         // up_off = up(off)*2
    up(featD, upFeat, NFEAT, 32, 32, 1.0f);                         // up_feat

    // ---------------- Level 2 (64x64) ----------------
    conv(nbr2, 64, ref2, 64,    P[ 9], P[10], offA, OCH, 64, 64, 1);   // oc1
    conv(offA, OCH, upOff, OCH, P[11], P[12], offB, OCH, 64, 64, 1);   // oc2(cat)
    conv(offB, OCH, nullptr, 0, P[13], P[14], offA, OCH, 64, 64, 1);   // oc3
    dcn(nbr2, offA, P[17], featD, 64, 64, 0);                          // dcn
    conv(featD, 64, upFeat, 64, P[15], P[16], featE, NFEAT, 64, 64, 1);// fc + leaky
    up(offA,  upOff,  OCH,   64, 64, 2.0f);
    up(featE, upFeat, NFEAT, 64, 64, 1.0f);

    // ---------------- Level 1 (128x128) ----------------
    conv(nbr1, 64, ref1, 64,    P[0], P[1], offA, OCH, 128, 128, 1);   // oc1
    conv(offA, OCH, upOff, OCH, P[2], P[3], offB, OCH, 128, 128, 1);   // oc2(cat)
    conv(offB, OCH, nullptr, 0, P[4], P[5], offA, OCH, 128, 128, 1);   // oc3
    dcn(nbr1, offA, P[8], featD, 128, 128, 0);                         // dcn
    conv(featD, 64, upFeat, 64, P[6], P[7], featE, NFEAT, 128, 128, 0);// fc (no act)

    // ---------------- Cascading ----------------
    conv(featE, 64, ref1, 64,   P[23], P[24], offA, OCH, 128, 128, 1); // cas1
    conv(offA, OCH, nullptr, 0, P[25], P[26], offB, OCH, 128, 128, 1); // cas2
    dcn(featE, offB, P[27], (float*)d_out, 128, 128, 1);               // leaky(dcn)
}
